// NUFFTLayerMultiChannelOneSided_59906203845043
// MI455X (gfx1250) — matrix operand order repaired and checked by reading, not compiled, not run on hardware
//
#include <hip/hip_runtime.h>
#include <hip/hip_bf16.h>
#include <math.h>

// ---------------- problem constants ----------------
#define NMESH   2001
#define NPART   1024
#define NBATCH  8
#define NCHAN   2
#define NKT     63          // K tiles: 63*32 = 2016 >= 2001

__device__ __host__ constexpr double PI_D = 3.14159265358979323846;
// TAU = 12 * (L / (2*pi*NMESH))^2, L = 1
__device__ constexpr float TAU_F =
    (float)(12.0 / ((2.0 * PI_D * 2001.0) * (2.0 * PI_D * 2001.0)));
__device__ constexpr float TWO_PI_F  = (float)(2.0 * PI_D);
__device__ constexpr float FOUR_PI_F = (float)(4.0 * PI_D);
__device__ constexpr float PI_F      = (float)PI_D;
__device__ constexpr float INV_N_F   = 1.0f / 2001.0f;

typedef __attribute__((ext_vector_type(16))) _Float16 v16h;
typedef __attribute__((ext_vector_type(8)))  float    v8f;

// ---------------------------------------------------------------------------
// Kernel 1: rho[b][n] = sum_p normalPer(x[b,p] - XGRID[n], sigma)
// ---------------------------------------------------------------------------
__global__ void k_rho(const float* __restrict__ x,
                      const float* __restrict__ sigma_var,
                      float* __restrict__ rho) {
    int idx = blockIdx.x * blockDim.x + threadIdx.x;
    if (idx >= NBATCH * NMESH) return;
    int b = idx / NMESH;
    int n = idx - b * NMESH;
    float xg = (float)n * INV_N_F;          // XGRID[n] = n / 2001
    float s = sigma_var[0];
    float inv2s2  = 0.5f / (s * s);
    float inv_nrm = 1.0f / (sqrtf(TWO_PI_F) * s);
    const float* xb = x + b * NPART;
    float acc = 0.0f;
    for (int p = 0; p < NPART; ++p) {
        float d  = xb[p] - xg;
        float dm = d - 1.0f;
        float dp = d + 1.0f;
        acc += __expf(-d  * d  * inv2s2)
             + __expf(-dm * dm * inv2s2)
             + __expf(-dp * dp * inv2s2);
    }
    rho[idx] = acc * inv_nrm;
}

// ---------------------------------------------------------------------------
// Kernel 2: real even convolution kernel equivalent to
//   ifft( ifftshift( mult * fftshift(fft(.)) ) ) / NMESH
// h[c][d] = (1/N^2)*( Mult_c(0) + 2*sum_{q=1..1000} Mult_c(q) cos(2 pi q d/N) )
// (multipliers depend only on k^2, hence even -> real circulant kernel)
// ---------------------------------------------------------------------------
__global__ void k_h(const float* __restrict__ shift0,
                    const float* __restrict__ shift1,
                    const float* __restrict__ amp0,
                    const float* __restrict__ amp1,
                    float* __restrict__ h) {
    int idx = blockIdx.x * blockDim.x + threadIdx.x;
    if (idx >= NCHAN * NMESH) return;
    int c = idx / NMESH;
    int d = idx - c * NMESH;
    float a0 = amp0[0], a1 = amp1[0];
    float s0 = 5.0f * shift0[0];
    float s1 = 5.0f * shift1[0];
    float s0sq = s0 * s0, s1sq = s1 * s1;
    float dscale = sqrtf(PI_F / TAU_F);     // sqrt(pi/tau)
    float acc = 0.0f;
    for (int q = 0; q <= NMESH / 2; ++q) {  // q = 0..1000
        float kk = TWO_PI_F * (float)q;
        float k2 = kk * kk;
        float deconv = dscale * __expf(k2 * TAU_F);
        float m;
        if (c == 0) {
            m = -a0 * FOUR_PI_F / (k2 + s0sq);
        } else {
            float t = 1.0f / (k2 + s1sq);
            m = a1 * FOUR_PI_F * t * t;
        }
        float mult = m * deconv;
        int r = (q * d) % NMESH;            // keep cos argument small & exact
        float cv = __cosf(TWO_PI_F * (float)r * INV_N_F);
        acc += (q == 0) ? mult : 2.0f * mult * cv;
    }
    h[idx] = acc * (INV_N_F * INV_N_F);     // 1/N (ifft) * 1/N (final /NMESH)
}

// ---------------------------------------------------------------------------
// Kernel 3: circular convolution conv[b][c][m] = sum_n rho[b][n]*h[c][(m-n)%N]
// ---------------------------------------------------------------------------
__global__ void k_conv(const float* __restrict__ rho,
                       const float* __restrict__ h,
                       float* __restrict__ conv) {
    int idx = blockIdx.x * blockDim.x + threadIdx.x;
    if (idx >= NBATCH * NCHAN * NMESH) return;
    int m  = idx % NMESH;
    int bc = idx / NMESH;
    int c  = bc & 1;
    int b  = bc >> 1;
    const float* r  = rho + b * NMESH;
    const float* hc = h   + c * NMESH;
    float acc = 0.0f;
    int d = m;                              // (m - n) mod N, n = 0
    for (int n = 0; n < NMESH; ++n) {
        acc = fmaf(r[n], hc[d], acc);
        d = (d == 0) ? (NMESH - 1) : (d - 1);
    }
    conv[idx] = acc;
}

// ---------------------------------------------------------------------------
// Kernel 3.5: pack conv into WMMA B-fragment order (f16).
// B K order chosen here (contiguous; A uses the same): lane l = (half<<4)|col,
// element i holds K_local = i + 16*half. Channels >= NCHAN and K >= NMESH
// are zero-padded, so the GEMM's padded region contributes exactly 0.
// bpack index: ((b*NKT + kt)*32 + lane) -> v16h (32 bytes, coalesced).
// ---------------------------------------------------------------------------
__global__ void k_pack_b(const float* __restrict__ conv,
                         v16h* __restrict__ bpack) {
    int tid = blockIdx.x * blockDim.x + threadIdx.x;   // 8*63*32 = 16128
    int lane = tid & 31;
    int kt   = (tid >> 5) % NKT;
    int b    = tid / (NKT * 32);
    int half = lane >> 4;
    int col  = lane & 15;
    v16h v;
    if (col < NCHAN) {
        const float* cv = conv + (b * NCHAN + col) * NMESH;
#pragma unroll
        for (int i = 0; i < 16; ++i) {
            int k = kt * 32 + i + 16 * half;
            v[i] = (_Float16)((k < NMESH) ? cv[k] : 0.0f);
        }
    } else {
#pragma unroll
        for (int i = 0; i < 16; ++i) v[i] = (_Float16)0.0f;
    }
    bpack[tid] = v;
}

// ---------------------------------------------------------------------------
// Kernel 4 (WMMA): fmm[b][p][c] = sum_n gauss(x[b,p]-XGRID[n]) * conv[b][c][n]
// Per-batch GEMM, A (gauss window) generated in-register, B from bpack.
// One wave32 per 16x16 output tile; 512 waves; EXEC all-1s around every WMMA.
// B stream software-pipelined one full k-step ahead (double buffer, unroll 2),
// plus a speculative global prefetch ~10 steps ahead for the B stream.
//
// A K order chosen here (contiguous; matches the B pack): lane = (half<<4)|m, element i holds
//   K = i + 16*half.  No K guard needed: padded K has B == 0.
// D layout: VGPR v, lane l -> row M = v + 8*(l>>4), col = l&15.
// ---------------------------------------------------------------------------
__global__ void k_fmm_wmma(const float* __restrict__ x,
                           const v16h* __restrict__ bpack,
                           float* __restrict__ out) {
    int wave = (blockIdx.x * blockDim.x + threadIdx.x) >> 5;  // 0..511
    int lane = threadIdx.x & 31;
    int b  = wave >> 6;                     // 0..7
    int pt = wave & 63;                     // particle tile
    int p0 = pt * 16;
    int mrow = lane & 15;
    int half = lane >> 4;
    int col  = lane & 15;                   // output column (channel if < NCHAN)

    float xp = x[b * NPART + p0 + mrow];
    const float inv4tau = 0.25f / TAU_F;
    const v16h* bp = bpack + b * NKT * 32 + lane;   // stride 32 fragments per kt

    // Gaussian gridding window A fragment for k-step kt, in WMMA A layout.
    auto makeA = [&](int kt) {
        v16h A;
        int kbase = kt * 32 + 16 * half;
#pragma unroll
        for (int i = 0; i < 16; ++i) {
            int ka = kbase + i;
            float d = xp - (float)ka * INV_N_F;
            A[i] = (_Float16)__expf(-d * d * inv4tau);
        }
        return A;
    };

    v8f acc = {};
    v16h Bc = bp[0];                        // fragment for kt = 0
    // NKT = 63 (odd): 31 double-steps cover kt = 0..61, epilogue does kt = 62.
    for (int kt = 0; kt < NKT - 1; kt += 2) {
        v16h Bn = bp[(kt + 1) * 32];        // issued a full step early
        __builtin_prefetch(&bp[(kt + 10) * 32], 0, 1);  // speculative, warms L2
        v16h A0 = makeA(kt);
        acc = __builtin_amdgcn_wmma_f32_16x16x32_f16(
                  false, A0, false, Bc, (short)0, acc, false, false);
        Bc = bp[(kt + 2) * 32];             // kt+2 <= 62: always in bounds
        v16h A1 = makeA(kt + 1);
        acc = __builtin_amdgcn_wmma_f32_16x16x32_f16(
                  false, A1, false, Bn, (short)0, acc, false, false);
    }
    {   // epilogue: kt = 62, fragment already resident in Bc
        v16h A2 = makeA(NKT - 1);
        acc = __builtin_amdgcn_wmma_f32_16x16x32_f16(
                  false, A2, false, Bc, (short)0, acc, false, false);
    }

    if (col < NCHAN) {
#pragma unroll
        for (int v = 0; v < 8; ++v) {
            int M = v + 8 * half;
            out[(b * NPART + p0 + M) * NCHAN + col] = acc[v];
        }
    }
}

// ---------------------------------------------------------------------------
// Launcher. Inputs: x(8*1024), sigma_var(1), shift0(1), shift1(1), amp0(1),
// amp1(1). Output: fmm (8*1024*2) f32.
// Workspace: rho(16008) + h(4002) + conv(32016) f32, pad to 32B, then
// bpack (8*63*32 v16h = 516096 bytes).
// ---------------------------------------------------------------------------
extern "C" void kernel_launch(void* const* d_in, const int* in_sizes, int n_in,
                              void* d_out, int out_size, void* d_ws, size_t ws_size,
                              hipStream_t stream) {
    const float* x         = (const float*)d_in[0];
    const float* sigma_var = (const float*)d_in[1];
    const float* shift0    = (const float*)d_in[2];
    const float* shift1    = (const float*)d_in[3];
    const float* amp0      = (const float*)d_in[4];
    const float* amp1      = (const float*)d_in[5];
    float* out = (float*)d_out;

    float* rho  = (float*)d_ws;             // 8   * 2001
    float* h    = rho + NBATCH * NMESH;     // 2   * 2001
    float* conv = h   + NCHAN  * NMESH;     // 8*2 * 2001
    // align bpack to 32 bytes
    size_t conv_end = (size_t)(NBATCH + NCHAN + NBATCH * NCHAN) * NMESH; // floats
    size_t bpack_off = (conv_end + 7) & ~(size_t)7;                      // 8-float = 32B
    v16h* bpack = (v16h*)((float*)d_ws + bpack_off);

    k_rho <<<(NBATCH * NMESH + 255) / 256, 256, 0, stream>>>(x, sigma_var, rho);
    k_h   <<<(NCHAN  * NMESH + 255) / 256, 256, 0, stream>>>(shift0, shift1, amp0, amp1, h);
    k_conv<<<(NBATCH * NCHAN * NMESH + 255) / 256, 256, 0, stream>>>(rho, h, conv);
    k_pack_b<<<(NBATCH * NKT * 32) / 256, 256, 0, stream>>>(conv, bpack);
    k_fmm_wmma<<<64, 256, 0, stream>>>(x, bpack, out);   // 512 waves, 1 tile each
}
